// MultiHeadAttention_6459630813642
// MI455X (gfx1250) — matrix operand order repaired and checked by reading, not compiled, not run on hardware
//
#include <hip/hip_runtime.h>

// MI455X / gfx1250 multi-head attention, f16 WMMA with f32 accumulate.
static constexpr int Bb = 2, Nn = 2048, Ee = 1024, Hh = 16, Dd = 64;

typedef __attribute__((ext_vector_type(16))) _Float16 v16h;
typedef __attribute__((ext_vector_type(8)))  _Float16 v8h;   // one b128 (16B)
typedef __attribute__((ext_vector_type(8)))  float    v8f;
typedef __attribute__((ext_vector_type(4)))  float    v4f;   // one b128 (16B)

union Frag16 { v16h v; v8h h[2]; };

// Order in-wave LDS store -> load (DS ops complete in-order per wave; this
// just stops the compiler from reordering across the staging point).
__device__ __forceinline__ void wave_fence_lds() {
  asm volatile("" ::: "memory");
  __builtin_amdgcn_wave_barrier();
  asm volatile("" ::: "memory");
}

// ---- fragment loaders -------------------------------------------------------
// A-operand (16x32 f16, M x K): per-lane row M = lane&15; element e holds
// K = koff + e, koff = 16*(lane>=16) [+32*kstep].
__device__ __forceinline__ v16h load_frag_a(const _Float16* base, int row,
                                            int rowStride, int koff) {
  Frag16 f;
  const _Float16* p = base + (size_t)row * rowStride + koff;
  f.h[0] = *(const v8h*)(p);
  f.h[1] = *(const v8h*)(p + 8);
  return f.v;
}
// B-operand (32x16 f16, K x N): per-lane column N = lane&15; lane group
// (lane>=16) covers K = kbase..kbase+15 contiguously (kbase = 16*group).
__device__ __forceinline__ v16h load_frag_b(const _Float16* base, int row,
                                            int rowStride, int kbase) {
  Frag16 f;
  const _Float16* p = base + (size_t)row * rowStride + kbase;
  f.h[0] = *(const v8h*)(p);
  f.h[1] = *(const v8h*)(p + 8);
  return f.v;
}

__device__ __forceinline__ v8f wmma_f16(v16h a, v16h b, v8f c) {
  return __builtin_amdgcn_wmma_f32_16x16x32_f16(false, a, false, b,
                                                (short)0, c, false, false);
}

// ---- kernel 0: cast Wo f32 -> f16 ------------------------------------------
__global__ __launch_bounds__(256) void cvt_f32_f16(const float* __restrict__ in,
                                                   _Float16* __restrict__ out,
                                                   int n) {
  int i = blockIdx.x * 256 + threadIdx.x;
  if (i < n) out[i] = (_Float16)in[i];
}

// ---- kernel 1: per-head projections ----------------------------------------
// which = blockIdx.z: 0->qh, 1->kh, 2->vhT(transposed [B,H,D,N]).
// qh[b,h,n,e] = sum_d X[b,n,h*D+d] * W[h,e,d]
__global__ __launch_bounds__(256) void proj_kernel(
    const float* __restrict__ q, const float* __restrict__ k,
    const float* __restrict__ v, const float* __restrict__ Wq,
    const float* __restrict__ Wk, const float* __restrict__ Wv,
    _Float16* __restrict__ qh, _Float16* __restrict__ kh,
    _Float16* __restrict__ vhT) {
  const int which = blockIdx.z;
  const int bh = blockIdx.y;                 // b*H + h
  const int b = bh / Hh, h = bh % Hh;
  const int n0 = blockIdx.x * 128;

  const float* X = (which == 0) ? q : (which == 1) ? k : v;
  const float* W = (which == 0) ? Wq : (which == 1) ? Wk : Wv;

  __shared__ _Float16 sX[128 * 72];          // padded stride 72 halves (144B)
  __shared__ _Float16 sW[64 * 72];
  __shared__ _Float16 sO[8][1536];           // per-wave output stage

  const int tid = threadIdx.x;
  for (int i = tid; i < 128 * 64; i += 256) {
    int r = i >> 6, c = i & 63;
    sX[r * 72 + c] = (_Float16)X[((size_t)b * Nn + n0 + r) * Ee + h * Dd + c];
  }
  for (int i = tid; i < 64 * 64; i += 256) {
    int r = i >> 6, c = i & 63;
    sW[r * 72 + c] = (_Float16)W[((size_t)h * Dd + r) * Dd + c];
  }
  __syncthreads();

  const int wave = tid >> 5, lane = tid & 31;
  const int hf = lane >> 4, ln = lane & 15;
  const int m0 = wave * 16;

  v16h a0 = load_frag_a(sX, m0 + ln, 72, 0 + hf * 16);
  v16h a1 = load_frag_a(sX, m0 + ln, 72, 32 + hf * 16);

  v8f c[4];
#pragma unroll
  for (int f = 0; f < 4; ++f) {
    const int e0 = f * 16;
    v16h b0 = load_frag_b(sW, e0 + ln, 72, 0 + hf * 16);
    v16h b1 = load_frag_b(sW, e0 + ln, 72, 32 + hf * 16);
    v8f acc = {};
    acc = wmma_f16(a0, b0, acc);
    c[f] = wmma_f16(a1, b1, acc);
  }

  _Float16* st = &sO[wave][0];
  if (which != 2) {
    // stage (n-local, e) row-major, stride 72
#pragma unroll
    for (int f = 0; f < 4; ++f)
#pragma unroll
      for (int r = 0; r < 8; ++r)
        st[(r + 8 * hf) * 72 + f * 16 + ln] = (_Float16)c[f][r];
    wave_fence_lds();
    // 16x64 tile is fully contiguous in qh/kh (row stride == D)
    _Float16* gout = ((which == 0) ? qh : kh) +
                     (size_t)bh * Nn * Dd + (size_t)(n0 + m0) * Dd;
#pragma unroll
    for (int i = 0; i < 4; ++i) {
      const int chunk = lane + 32 * i;             // 128 chunks of 8 halves
      const int row = chunk >> 3, col = (chunk & 7) * 8;
      *(v8h*)(gout + chunk * 8) = *(const v8h*)(st + row * 72 + col);
    }
  } else {
    // stage transposed (e, n-local), stride 24
#pragma unroll
    for (int f = 0; f < 4; ++f)
#pragma unroll
      for (int r = 0; r < 8; ++r)
        st[(f * 16 + ln) * 24 + r + 8 * hf] = (_Float16)c[f][r];
    wave_fence_lds();
    _Float16* gout = vhT + (size_t)bh * Dd * Nn + (n0 + m0);
#pragma unroll
    for (int i = 0; i < 4; ++i) {
      const int chunk = lane + 32 * i;             // 64 rows x 2 chunks
      const int e = chunk >> 1, col = (chunk & 1) * 8;
      *(v8h*)(gout + (size_t)e * Nn + col) = *(const v8h*)(st + e * 24 + col);
    }
  }
}

// ---- kernel 2: flash attention ---------------------------------------------
// Wave = 16 queries x full D=64. Loop over key blocks of 32.
// Row-sum of P is folded into the PV WMMA via a constant ones-column
// B-fragment: o5[:,0] accumulates sum_k P[q,k] with the same alpha rescale
// as the output accumulators, so no per-iteration sum shuffles are needed.
__global__ __launch_bounds__(256) void attn_kernel(
    const _Float16* __restrict__ qh, const _Float16* __restrict__ kh,
    const _Float16* __restrict__ vhT, const int* __restrict__ mask,
    _Float16* __restrict__ attn) {
  const int bh = blockIdx.y;
  const int b = bh / Hh, h = bh % Hh;
  const int tid = threadIdx.x, wave = tid >> 5, lane = tid & 31;
  const int hf = lane >> 4, ln = lane & 15;
  const int q0 = blockIdx.x * 128 + wave * 16;

  const _Float16* Q = qh + (size_t)bh * Nn * Dd;
  const _Float16* K = kh + (size_t)bh * Nn * Dd;
  const _Float16* VT = vhT + (size_t)bh * Dd * Nn;

  __shared__ _Float16 sP[8][1152];           // P tile (stride 40) + out stage
  _Float16* myP = &sP[wave][0];

  v16h aq0 = load_frag_a(Q, q0 + ln, Dd, 0 + hf * 16);
  v16h aq1 = load_frag_a(Q, q0 + ln, Dd, 32 + hf * 16);

  // B-layout constant: column 0 = all ones (lanes with ln==0 hold 1s).
  v16h bOnes;
  {
    const _Float16 on = (ln == 0) ? (_Float16)1.0f : (_Float16)0.0f;
#pragma unroll
    for (int i = 0; i < 16; ++i) bOnes[i] = on;
  }

  float bias[8], m[8];
  v8f o[4] = {};
  v8f o5 = {};                               // column 0 = running row-sum l
#pragma unroll
  for (int r = 0; r < 8; ++r) {
    const int qi = q0 + r + 8 * hf;
    bias[r] = mask[(size_t)b * Nn + qi] ? 0.0f : -1e28f;  // query-row mask
    m[r] = -3.0e38f;
  }
  const float scale = 0.03125f;  // 1/sqrt(E) = 1/32

  for (int k0 = 0; k0 < Nn; k0 += 32) {
    if (k0 + 32 < Nn) {  // prefetch next K/V tiles (global_prefetch_b8)
      __builtin_prefetch(K + (size_t)(k0 + 32 + lane) * Dd, 0, 0);
      __builtin_prefetch(VT + (size_t)lane * Nn + k0 + 32, 0, 0);
      __builtin_prefetch(VT + (size_t)(32 + lane) * Nn + k0 + 32, 0, 0);
    }
    // S = Q·K^T  (16x32 tile = two 16x16 C frags)
    v8f s0 = {}, s1 = {};
    {
      v16h bk;
      bk = load_frag_b(K, k0 + ln, Dd, 0 + hf * 16);       s0 = wmma_f16(aq0, bk, s0);
      bk = load_frag_b(K, k0 + ln, Dd, 32 + hf * 16);      s0 = wmma_f16(aq1, bk, s0);
      bk = load_frag_b(K, k0 + 16 + ln, Dd, 0 + hf * 16);  s1 = wmma_f16(aq0, bk, s1);
      bk = load_frag_b(K, k0 + 16 + ln, Dd, 32 + hf * 16); s1 = wmma_f16(aq1, bk, s1);
    }
    // online softmax; row r lives across the 16 lanes of this half
#pragma unroll
    for (int r = 0; r < 8; ++r) {
      float x0 = s0[r] * scale + bias[r];
      float x1 = s1[r] * scale + bias[r];
      float vmax = fmaxf(x0, x1);
      vmax = fmaxf(vmax, __shfl_xor(vmax, 1, 16));
      vmax = fmaxf(vmax, __shfl_xor(vmax, 2, 16));
      vmax = fmaxf(vmax, __shfl_xor(vmax, 4, 16));
      vmax = fmaxf(vmax, __shfl_xor(vmax, 8, 16));
      const float mn = fmaxf(m[r], vmax);
      const float alpha = __expf(m[r] - mn);
      m[r] = mn;
      const float p0 = __expf(x0 - mn);
      const float p1 = __expf(x1 - mn);
#pragma unroll
      for (int f = 0; f < 4; ++f) o[f][r] = o[f][r] * alpha;
      o5[r] = o5[r] * alpha;
      const int row = r + 8 * hf;
      myP[row * 40 + ln] = (_Float16)p0;      // cols 0..15
      myP[row * 40 + 16 + ln] = (_Float16)p1; // cols 16..31
    }
    wave_fence_lds();
    v16h ap = load_frag_a(myP, ln, 40, hf * 16);   // P in A layout (K=32)
#pragma unroll
    for (int f = 0; f < 4; ++f) {
      v16h bv = load_frag_b(VT, f * 16 + ln, Nn, k0 + hf * 16);
      o[f] = wmma_f16(ap, bv, o[f]);
    }
    o5 = wmma_f16(ap, bOnes, o5);              // row-sum accumulation
  }

  // finalize: broadcast l from lane 0 of each half-group, /l, stage, store
  float l[8];
#pragma unroll
  for (int r = 0; r < 8; ++r) l[r] = __shfl(o5[r], 0, 16);

  wave_fence_lds();
#pragma unroll
  for (int f = 0; f < 4; ++f)
#pragma unroll
    for (int r = 0; r < 8; ++r)
      myP[(r + 8 * hf) * 72 + f * 16 + ln] = (_Float16)(o[f][r] / l[r]);
  wave_fence_lds();
#pragma unroll
  for (int i = 0; i < 4; ++i) {
    const int chunk = lane + 32 * i;               // 16 rows x 8 chunks
    const int row = chunk >> 3, col = (chunk & 7) * 8;
    *(v8h*)(attn + ((size_t)b * Nn + q0 + row) * Ee + h * Dd + col) =
        *(const v8h*)(myP + row * 72 + col);
  }
}

// ---- kernel 3: output projection  out = attn @ Wo^T  (f32 out) -------------
__global__ __launch_bounds__(256) void out_proj_kernel(
    const _Float16* __restrict__ attn, const _Float16* __restrict__ wo,
    float* __restrict__ out) {
  const int tid = threadIdx.x, wave = tid >> 5, lane = tid & 31;
  const int hf = lane >> 4, ln = lane & 15;
  const int r0 = blockIdx.x * 128 + wave * 16;   // rows over B*N = 4096
  const int c0 = blockIdx.y * 64;                // cols over E

  __shared__ float sOf[8][16 * 68];              // padded f32 stage

  v8f acc[4] = {};
  for (int kk = 0; kk < Ee; kk += 32) {
    if (kk + 32 < Ee)
      __builtin_prefetch(wo + (size_t)(c0 + (lane & 31)) * Ee + kk + 32, 0, 0);
    v16h a = load_frag_a(attn, r0 + ln, Ee, kk + hf * 16);
#pragma unroll
    for (int f = 0; f < 4; ++f) {
      v16h bb = load_frag_b(wo, c0 + f * 16 + ln, Ee, kk + hf * 16);
      acc[f] = wmma_f16(a, bb, acc[f]);
    }
  }
  float* st = &sOf[wave][0];
#pragma unroll
  for (int f = 0; f < 4; ++f)
#pragma unroll
    for (int r = 0; r < 8; ++r)
      st[(r + 8 * hf) * 68 + f * 16 + ln] = acc[f][r];
  wave_fence_lds();
#pragma unroll
  for (int i = 0; i < 8; ++i) {
    const int chunk = lane + 32 * i;               // 16 rows x 16 chunks (4 f32)
    const int row = chunk >> 4, col = (chunk & 15) * 4;
    *(v4f*)(out + (size_t)(r0 + row) * Ee + c0 + col) =
        *(const v4f*)(st + row * 68 + col);
  }
}

// ---- host -------------------------------------------------------------------
extern "C" void kernel_launch(void* const* d_in, const int* in_sizes, int n_in,
                              void* d_out, int out_size, void* d_ws,
                              size_t ws_size, hipStream_t stream) {
  const float* q   = (const float*)d_in[0];
  const float* k   = (const float*)d_in[1];
  const float* v   = (const float*)d_in[2];
  const int*   msk = (const int*)d_in[3];
  const float* Wq  = (const float*)d_in[4];
  const float* Wk  = (const float*)d_in[5];
  const float* Wv  = (const float*)d_in[6];
  const float* Wo  = (const float*)d_in[7];

  const size_t BHND = (size_t)Bb * Hh * Nn * Dd;   // 4,194,304
  const size_t BNE  = (size_t)Bb * Nn * Ee;        // 4,194,304
  const size_t need = (3 * BHND + BNE + (size_t)Ee * Ee) * sizeof(_Float16);
  if (ws_size < need) return;

  _Float16* ws   = (_Float16*)d_ws;
  _Float16* qh   = ws;
  _Float16* kh   = qh + BHND;
  _Float16* vhT  = kh + BHND;
  _Float16* attn = vhT + BHND;
  _Float16* woh  = attn + BNE;

  cvt_f32_f16<<<(Ee * Ee + 255) / 256, 256, 0, stream>>>(Wo, woh, Ee * Ee);
  proj_kernel<<<dim3(Nn / 128, Bb * Hh, 3), 256, 0, stream>>>(
      q, k, v, Wq, Wk, Wv, qh, kh, vhT);
  attn_kernel<<<dim3(Nn / 128, Bb * Hh), 256, 0, stream>>>(
      qh, kh, vhT, msk, attn);
  out_proj_kernel<<<dim3((Bb * Nn) / 128, Ee / 64), 256, 0, stream>>>(
      attn, woh, (float*)d_out);
}
